// TensorBase_46222438039976
// MI455X (gfx1250) — matrix operand order repaired and checked by reading, not compiled, not run on hardware
//
#include <hip/hip_runtime.h>
#include <stdint.h>

// ---------------------------------------------------------------------------
// TensoRF-style renderer for MI455X (gfx1250, wave32, WMMA, async LDS DMA)
// ---------------------------------------------------------------------------

typedef __attribute__((ext_vector_type(16))) _Float16 v16h;
typedef __attribute__((ext_vector_type(8)))  float    v8f;

#define R_RAYS   1024
#define N_SAMP   256
#define NPTS     (R_RAYS * N_SAMP)
#define GRID_N   256
#define D_COMP   8
#define A_COMP   24
#define APP_DIM  27
#define FEAT_C   128
#define IN_MLP   174       // 27 + 3 + 108 + 36
#define K1_PAD   192       // IN_MLP padded to multiple of 32
#define CHUNK_RAYS 256
#define CHUNK_PTS  (CHUNK_RAYS * N_SAMP)   // 65536

__device__ __forceinline__ float softplusf(float x) {
    return (x > 20.f) ? x : log1pf(__expf(x));
}
__device__ __forceinline__ int clampi(int v, int lo, int hi) {
    return v < lo ? lo : (v > hi ? hi : v);
}
__device__ __forceinline__ float clampf(float v, float lo, float hi) {
    return fminf(fmaxf(v, lo), hi);
}

// --- gfx1250 async global<->LDS DMA (ASYNCcnt-tracked, no VGPR round trip) ---
__device__ __forceinline__ void async_load_to_lds_b128(void* lds_ptr, const void* gptr) {
    unsigned lds = (unsigned)(unsigned long long)lds_ptr;   // low 32b = LDS offset
    unsigned long long ga = (unsigned long long)gptr;
    asm volatile("global_load_async_to_lds_b128 %0, %1, off"
                 :: "v"(lds), "v"(ga) : "memory");
}
__device__ __forceinline__ void async_store_from_lds_b128(void* gptr, const void* lds_ptr) {
    unsigned lds = (unsigned)(unsigned long long)lds_ptr;
    unsigned long long ga = (unsigned long long)gptr;
    asm volatile("global_store_async_from_lds_b128 %0, %1, off"
                 :: "v"(ga), "v"(lds) : "memory");
}
__device__ __forceinline__ void wait_asynccnt0() {
    asm volatile("s_wait_asynccnt 0x0" ::: "memory");
}

// ---------------------------------------------------------------------------
// Kernel 0: convert + transpose MLP weights to f16  (W1 -> [128][192] zero-pad,
// W2 -> [128][128]).  Tiny, runs once per launch.
// ---------------------------------------------------------------------------
__global__ void convert_weights_kernel(const float* __restrict__ W1,
                                       const float* __restrict__ W2,
                                       _Float16* __restrict__ w1hT,
                                       _Float16* __restrict__ w2hT) {
    int t = blockIdx.x * blockDim.x + threadIdx.x;
    const int N1 = FEAT_C * K1_PAD;      // 24576
    const int N2 = FEAT_C * FEAT_C;      // 16384
    if (t < N1) {
        int n = t / K1_PAD, k = t % K1_PAD;
        w1hT[t] = (k < IN_MLP) ? (_Float16)W1[k * FEAT_C + n] : (_Float16)0.f;
    } else if (t < N1 + N2) {
        int t2 = t - N1;
        int n = t2 / FEAT_C, k = t2 % FEAT_C;
        w2hT[t2] = (_Float16)W2[k * FEAT_C + n];
    }
}

// ---------------------------------------------------------------------------
// Kernel 1: density sampling -> alpha per sample point
// ---------------------------------------------------------------------------
__global__ void density_kernel(const float* __restrict__ rays,
                               const float* __restrict__ dplanes,
                               const float* __restrict__ dlines,
                               float* __restrict__ alphaOut) {
    int i = blockIdx.x * blockDim.x + threadIdx.x;
    if (i >= NPTS) return;
    int r = i >> 8, s = i & 255;
    const float z = 0.05f + (float)s * 0.01f;          // linspace(0.05,2.6,256)
    float ox = rays[r * 6 + 0], oy = rays[r * 6 + 1], oz = rays[r * 6 + 2];
    float dx = rays[r * 6 + 3], dy = rays[r * 6 + 4], dz = rays[r * 6 + 5];
    float px = ox + dx * z, py = oy + dy * z, pz = oz + dz * z;
    bool valid = (px >= -1.5f) & (px <= 1.5f) & (py >= -1.5f) & (py <= 1.5f)
               & (pz >= -1.5f) & (pz <= 1.5f);
    float X = (px + 1.5f) * (2.f / 3.f) - 1.f;
    float Y = (py + 1.5f) * (2.f / 3.f) - 1.f;
    float Z = (pz + 1.5f) * (2.f / 3.f) - 1.f;

    const float us[3] = {X, X, Y};
    const float vs[3] = {Y, Z, Z};
    const float ls[3] = {Z, Y, X};

    float sig = 0.f;
#pragma unroll
    for (int p = 0; p < 3; ++p) {
        float fx = (us[p] + 1.f) * 0.5f * 255.f;
        float fy = (vs[p] + 1.f) * 0.5f * 255.f;
        float fl = (ls[p] + 1.f) * 0.5f * 255.f;
        int x0 = clampi((int)floorf(fx), 0, 254);
        int y0 = clampi((int)floorf(fy), 0, 254);
        int l0 = clampi((int)floorf(fl), 0, 254);
        float wx = clampf(fx - (float)x0, 0.f, 1.f);
        float wy = clampf(fy - (float)y0, 0.f, 1.f);
        float wl = clampf(fl - (float)l0, 0.f, 1.f);
        const float* pl = dplanes + (size_t)p * D_COMP * GRID_N * GRID_N;
        const float* ln = dlines  + (size_t)p * D_COMP * GRID_N;
        int b00 = y0 * GRID_N + x0;
#pragma unroll
        for (int c = 0; c < D_COMP; ++c) {
            const float* pc = pl + (size_t)c * GRID_N * GRID_N;
            float v00 = pc[b00], v01 = pc[b00 + 1];
            float v10 = pc[b00 + GRID_N], v11 = pc[b00 + GRID_N + 1];
            float pv = v00 * (1.f - wx) * (1.f - wy) + v01 * wx * (1.f - wy)
                     + v10 * (1.f - wx) * wy + v11 * wx * wy;
            float lv = ln[c * GRID_N + l0] * (1.f - wl) + ln[c * GRID_N + l0 + 1] * wl;
            sig += pv * lv;
        }
    }
    float sigma = valid ? softplusf(sig - 10.f) : 0.f;
    alphaOut[i] = 1.f - __expf(-sigma * 0.25f);  // dist*DISTANCE_SCALE = 0.01*25
}

// ---------------------------------------------------------------------------
// Kernel 2: per-ray transmittance scan -> weights
// ---------------------------------------------------------------------------
__global__ void scan_kernel(const float* __restrict__ alpha,
                            float* __restrict__ weights) {
    int r = blockIdx.x * blockDim.x + threadIdx.x;
    if (r >= R_RAYS) return;
    float T = 1.f;
    for (int s = 0; s < N_SAMP; ++s) {
        float a = alpha[r * N_SAMP + s];
        weights[r * N_SAMP + s] = a * T;
        T *= (1.f - a + 1e-10f);
    }
}

// ---------------------------------------------------------------------------
// Kernel 3: appearance sampling + basis matvec + positional encodings
// Emits f16 MLP-input rows [CHUNK_PTS][192] (cols 174..191 zeroed)
// ---------------------------------------------------------------------------
__global__ void feat_kernel(const float* __restrict__ rays,
                            const float* __restrict__ aplanes,
                            const float* __restrict__ alines,
                            const float* __restrict__ basisW,
                            _Float16* __restrict__ Aout,
                            int ptBase) {
    __shared__ float bsm[3 * A_COMP * APP_DIM];  // 72*27 = 1944 floats
    for (int t = threadIdx.x; t < 3 * A_COMP * APP_DIM; t += blockDim.x)
        bsm[t] = basisW[t];
    __syncthreads();

    int i = blockIdx.x * blockDim.x + threadIdx.x;   // chunk-local point
    if (i >= CHUNK_PTS) return;
    int g = ptBase + i;
    int r = g >> 8, s = g & 255;
    const float z = 0.05f + (float)s * 0.01f;
    float ox = rays[r * 6 + 0], oy = rays[r * 6 + 1], oz = rays[r * 6 + 2];
    float dx = rays[r * 6 + 3], dy = rays[r * 6 + 4], dz = rays[r * 6 + 5];
    float px = ox + dx * z, py = oy + dy * z, pz = oz + dz * z;
    float X = (px + 1.5f) * (2.f / 3.f) - 1.f;
    float Y = (py + 1.5f) * (2.f / 3.f) - 1.f;
    float Z = (pz + 1.5f) * (2.f / 3.f) - 1.f;

    const float us[3] = {X, X, Y};
    const float vs[3] = {Y, Z, Z};
    const float ls[3] = {Z, Y, X};

    float acc[APP_DIM];
#pragma unroll
    for (int k = 0; k < APP_DIM; ++k) acc[k] = 0.f;

    for (int p = 0; p < 3; ++p) {
        float fx = (us[p] + 1.f) * 0.5f * 255.f;
        float fy = (vs[p] + 1.f) * 0.5f * 255.f;
        float fl = (ls[p] + 1.f) * 0.5f * 255.f;
        int x0 = clampi((int)floorf(fx), 0, 254);
        int y0 = clampi((int)floorf(fy), 0, 254);
        int l0 = clampi((int)floorf(fl), 0, 254);
        float wx = clampf(fx - (float)x0, 0.f, 1.f);
        float wy = clampf(fy - (float)y0, 0.f, 1.f);
        float wl = clampf(fl - (float)l0, 0.f, 1.f);
        const float* pl = aplanes + (size_t)p * A_COMP * GRID_N * GRID_N;
        const float* ln = alines  + (size_t)p * A_COMP * GRID_N;
        int b00 = y0 * GRID_N + x0;
        for (int c = 0; c < A_COMP; ++c) {
            const float* pc = pl + (size_t)c * GRID_N * GRID_N;
            float v00 = pc[b00], v01 = pc[b00 + 1];
            float v10 = pc[b00 + GRID_N], v11 = pc[b00 + GRID_N + 1];
            float pv = v00 * (1.f - wx) * (1.f - wy) + v01 * wx * (1.f - wy)
                     + v10 * (1.f - wx) * wy + v11 * wx * wy;
            float lv = ln[c * GRID_N + l0] * (1.f - wl) + ln[c * GRID_N + l0 + 1] * wl;
            float w = pv * lv;
            const float* bw = &bsm[(p * A_COMP + c) * APP_DIM];
#pragma unroll
            for (int k = 0; k < APP_DIM; ++k) acc[k] = fmaf(w, bw[k], acc[k]);
        }
    }

    _Float16* Ar = Aout + (size_t)i * K1_PAD;
#pragma unroll
    for (int k = 0; k < APP_DIM; ++k) Ar[k] = (_Float16)acc[k];
    Ar[27] = (_Float16)dx; Ar[28] = (_Float16)dy; Ar[29] = (_Float16)dz;
    // PE(feat, 2 freqs): sin @ [30..83], cos @ [84..137], dim-major, freq-minor
#pragma unroll
    for (int j = 0; j < APP_DIM; ++j) {
#pragma unroll
        for (int f = 0; f < 2; ++f) {
            float ang = acc[j] * (float)(1 << f);
            Ar[30 + j * 2 + f] = (_Float16)__sinf(ang);
            Ar[84 + j * 2 + f] = (_Float16)__cosf(ang);
        }
    }
    // PE(vd, 6 freqs): sin @ [138..155], cos @ [156..173]
    const float vd[3] = {dx, dy, dz};
#pragma unroll
    for (int j = 0; j < 3; ++j) {
#pragma unroll
        for (int f = 0; f < 6; ++f) {
            float ang = vd[j] * (float)(1 << f);
            Ar[138 + j * 6 + f] = (_Float16)__sinf(ang);
            Ar[156 + j * 6 + f] = (_Float16)__cosf(ang);
        }
    }
#pragma unroll
    for (int k = IN_MLP; k < K1_PAD; ++k) Ar[k] = (_Float16)0.f;
}

// ---------------------------------------------------------------------------
// Kernel 4/5: WMMA GEMM + bias + relu.
//   C[M,128] = relu(A[M,K] @ B[K,128] + bias),  B given transposed [128][K] f16
//   Block = 256 threads = 8 waves; block tile 128M x 128N; wave tile 32M x 64N.
//   A tiles streamed via global_load_async_to_lds_b128 overlapped with WMMAs;
//   C tile staged in LDS and written with global_store_async_from_lds_b128.
// ---------------------------------------------------------------------------
union HFrag { uint4 q[2]; v16h h; };

#define CSM_STRIDE 136                        // halfs; rows 16B-aligned, bank-safe
#define CSM_BYTES  (128 * CSM_STRIDE * 2)     // 34816

template <int K>
__global__ void gemm_relu_kernel(const _Float16* __restrict__ A,
                                 const _Float16* __restrict__ BT,  // [128][K]
                                 const float* __restrict__ bias,
                                 _Float16* __restrict__ out) {
    constexpr int BSM_BYTES  = FEAT_C * K * 2;          // 48K / 32K
    constexpr int ASM_BYTES  = 128 * 40 * 2;            // 10K (stride-40 halfs)
    constexpr int LIVE_BYTES = BSM_BYTES + ASM_BYTES;
    constexpr int SMEM_BYTES = (LIVE_BYTES > CSM_BYTES) ? LIVE_BYTES : CSM_BYTES;
    __shared__ __align__(16) char smem[SMEM_BYTES];
    _Float16* Bsm = (_Float16*)smem;                    // weights, live all K loop
    _Float16* Asm = (_Float16*)(smem + BSM_BYTES);      // A chunk [128][40]
    _Float16* Csm = (_Float16*)smem;                    // epilogue staging (aliases)

    const int blockRow = blockIdx.x * 128;

    // ---- weight matrix -> LDS once (tiny, L2-resident) ----
    {
        const int totQ = FEAT_C * K / 8;
        const uint4* src = (const uint4*)BT;
        uint4* dst = (uint4*)Bsm;
        for (int t = threadIdx.x; t < totQ; t += 256) dst[t] = src[t];
    }
    // ---- kick off async DMA of A chunk 0 while B copy drains ----
#pragma unroll
    for (int ss = 0; ss < 2; ++ss) {
        int s2 = threadIdx.x + ss * 256;
        int row = s2 >> 2, seg = s2 & 3;
        const _Float16* gp = A + (size_t)(blockRow + row) * K + seg * 8;
        async_load_to_lds_b128(&Asm[row * 40 + seg * 8], gp);
    }
    __syncthreads();   // B copy visible (dscnt waited inside)

    const int lane = threadIdx.x & 31;
    const int wave = threadIdx.x >> 5;
    const int mRow = (wave >> 1) * 32;       // 0,32,64,96
    const int nCol = (wave & 1) * 64;        // 0,64
    v8f acc[2][4] = {};

    const int aKbase = (lane < 16) ? 0 : 8;  // ISA A-fragment K layout
    const int aRow   = (lane & 15);
    const int bHalf  = (lane >> 4) * 8;      // ISA B-fragment K layout
    const int bN     = nCol + (lane & 15);

    for (int kc = 0; kc < K / 32; ++kc) {
        wait_asynccnt0();                    // A chunk kc landed in LDS
        __syncthreads();

        // ---- A fragments (2 M-subtiles) ----
        HFrag af[2];
#pragma unroll
        for (int mt = 0; mt < 2; ++mt) {
            const _Float16* p = &Asm[(mRow + mt * 16 + aRow) * 40 + aKbase];
            af[mt].q[0] = *(const uint4*)(p);
            af[mt].q[1] = *(const uint4*)(p + 16);
        }
        // ---- B fragments (4 N-subtiles) ----
        HFrag bf[4];
#pragma unroll
        for (int nt = 0; nt < 4; ++nt) {
            const _Float16* p = &Bsm[(bN + nt * 16) * K + kc * 32 + bHalf];
            bf[nt].q[0] = *(const uint4*)(p);
            bf[nt].q[1] = *(const uint4*)(p + 16);
        }
        __syncthreads();   // all waves done reading Asm (dscnt waited inside)

        // ---- overlap: DMA next A chunk while WMMAs run ----
        if (kc + 1 < K / 32) {
#pragma unroll
            for (int ss = 0; ss < 2; ++ss) {
                int s2 = threadIdx.x + ss * 256;
                int row = s2 >> 2, seg = s2 & 3;
                const _Float16* gp =
                    A + (size_t)(blockRow + row) * K + (kc + 1) * 32 + seg * 8;
                async_load_to_lds_b128(&Asm[row * 40 + seg * 8], gp);
                if (kc + 2 < K / 32)
                    __builtin_prefetch(gp + 32, 0, 1);   // warm L2 for kc+2
            }
        }
        // ---- 8 WMMAs ----
#pragma unroll
        for (int mt = 0; mt < 2; ++mt)
#pragma unroll
            for (int nt = 0; nt < 4; ++nt)
                acc[mt][nt] = __builtin_amdgcn_wmma_f32_16x16x32_f16(
                    false, af[mt].h, false, bf[nt].h,
                    (short)0, acc[mt][nt], false, false);
    }

    // ---- epilogue: bias + relu -> f16, stage in LDS (Bsm/Asm are dead) ----
#pragma unroll
    for (int mt = 0; mt < 2; ++mt) {
#pragma unroll
        for (int nt = 0; nt < 4; ++nt) {
            int n = nCol + nt * 16 + (lane & 15);
            float bi = bias[n];
            int m0 = mRow + mt * 16 + (lane >> 4) * 8;
#pragma unroll
            for (int rr = 0; rr < 8; ++rr) {
                float x = fmaxf(acc[mt][nt][rr] + bi, 0.f);
                Csm[(m0 + rr) * CSM_STRIDE + n] = (_Float16)x;
            }
        }
    }
    __syncthreads();
    // ---- coalesced async LDS -> global stores (b128) ----
    for (int t = threadIdx.x; t < 2048; t += 256) {
        int row = t >> 4, seg = t & 15;
        async_store_from_lds_b128(out + (size_t)(blockRow + row) * FEAT_C + seg * 8,
                                  &Csm[row * CSM_STRIDE + seg * 8]);
    }
    wait_asynccnt0();
}

// ---------------------------------------------------------------------------
// Kernel 6: final layer (128x3) + sigmoid + masked weighted composite per ray
// One block per ray, one thread per sample.
// ---------------------------------------------------------------------------
__global__ void final_kernel(const _Float16* __restrict__ h2,
                             const float* __restrict__ W3,
                             const float* __restrict__ b3,
                             const float* __restrict__ weights,
                             float* __restrict__ outRGB,
                             int rayBase) {
    __shared__ float w3s[FEAT_C * 3];
    __shared__ float b3s[3];
    __shared__ float red[4][N_SAMP];

    for (int t = threadIdx.x; t < FEAT_C * 3; t += blockDim.x) w3s[t] = W3[t];
    if (threadIdx.x < 3) b3s[threadIdx.x] = b3[threadIdx.x];
    __syncthreads();

    int ray = rayBase + blockIdx.x;
    int s = threadIdx.x;
    int li = blockIdx.x * N_SAMP + s;          // chunk-local point index
    int gi = ray * N_SAMP + s;                 // global sample index

    const _Float16* hp = h2 + (size_t)li * FEAT_C;
    float c0 = b3s[0], c1 = b3s[1], c2 = b3s[2];
#pragma unroll 8
    for (int j = 0; j < FEAT_C; ++j) {
        float hv = (float)hp[j];
        c0 = fmaf(hv, w3s[j * 3 + 0], c0);
        c1 = fmaf(hv, w3s[j * 3 + 1], c1);
        c2 = fmaf(hv, w3s[j * 3 + 2], c2);
    }
    float s0 = 1.f / (1.f + __expf(-c0));
    float s1 = 1.f / (1.f + __expf(-c1));
    float s2 = 1.f / (1.f + __expf(-c2));
    float w = weights[gi];
    float m = (w > 1e-4f) ? 1.f : 0.f;
    red[0][s] = w * m * s0;
    red[1][s] = w * m * s1;
    red[2][s] = w * m * s2;
    red[3][s] = w;
    __syncthreads();

    for (int st = N_SAMP / 2; st > 0; st >>= 1) {
        if (s < st) {
            red[0][s] += red[0][s + st];
            red[1][s] += red[1][s + st];
            red[2][s] += red[2][s + st];
            red[3][s] += red[3][s + st];
        }
        __syncthreads();
    }
    if (s == 0) {
        float accw = red[3][0];
        float bg = 1.f - accw;
        outRGB[ray * 3 + 0] = clampf(red[0][0] + bg, 0.f, 1.f);
        outRGB[ray * 3 + 1] = clampf(red[1][0] + bg, 0.f, 1.f);
        outRGB[ray * 3 + 2] = clampf(red[2][0] + bg, 0.f, 1.f);
    }
}

// ---------------------------------------------------------------------------
// Launch
// ---------------------------------------------------------------------------
extern "C" void kernel_launch(void* const* d_in, const int* in_sizes, int n_in,
                              void* d_out, int out_size, void* d_ws, size_t ws_size,
                              hipStream_t stream) {
    const float* rays    = (const float*)d_in[0];
    const float* dplanes = (const float*)d_in[1];
    const float* dlines  = (const float*)d_in[2];
    const float* aplanes = (const float*)d_in[3];
    const float* alines  = (const float*)d_in[4];
    const float* basisW  = (const float*)d_in[5];
    const float* W1      = (const float*)d_in[6];
    const float* b1      = (const float*)d_in[7];
    const float* W2      = (const float*)d_in[8];
    const float* b2      = (const float*)d_in[9];
    const float* W3      = (const float*)d_in[10];
    const float* b3      = (const float*)d_in[11];
    float* out           = (float*)d_out;

    // workspace layout (all sizes multiples of 256B)
    const size_t SZ_W1H = (size_t)FEAT_C * K1_PAD * 2;     //   48 KB
    const size_t SZ_W2H = (size_t)FEAT_C * FEAT_C * 2;     //   32 KB
    const size_t SZ_A   = (size_t)NPTS * 4;                //    1 MB (alpha)
    const size_t SZ_W   = (size_t)NPTS * 4;                //    1 MB (weights)
    const size_t SZ_AIN = (size_t)CHUNK_PTS * K1_PAD * 2;  //   24 MB
    const size_t SZ_H   = (size_t)CHUNK_PTS * FEAT_C * 2;  //   16 MB
    uint8_t* base = (uint8_t*)d_ws;
    _Float16* w1hT  = (_Float16*)(base);
    _Float16* w2hT  = (_Float16*)(base + SZ_W1H);
    float*    alpha = (float*)   (base + SZ_W1H + SZ_W2H);
    float*    wgt   = (float*)   (base + SZ_W1H + SZ_W2H + SZ_A);
    _Float16* a_in  = (_Float16*)(base + SZ_W1H + SZ_W2H + SZ_A + SZ_W);
    _Float16* h1    = (_Float16*)(base + SZ_W1H + SZ_W2H + SZ_A + SZ_W + SZ_AIN);
    _Float16* h2    = (_Float16*)(base + SZ_W1H + SZ_W2H + SZ_A + SZ_W + SZ_AIN + SZ_H);
    if (ws_size < SZ_W1H + SZ_W2H + SZ_A + SZ_W + SZ_AIN + 2 * SZ_H) return;

    convert_weights_kernel<<<160, 256, 0, stream>>>(W1, W2, w1hT, w2hT);
    density_kernel<<<NPTS / 256, 256, 0, stream>>>(rays, dplanes, dlines, alpha);
    scan_kernel<<<R_RAYS / 256, 256, 0, stream>>>(alpha, wgt);

    for (int c = 0; c < R_RAYS / CHUNK_RAYS; ++c) {
        int ptBase = c * CHUNK_PTS;
        int rayBase = c * CHUNK_RAYS;
        feat_kernel<<<CHUNK_PTS / 256, 256, 0, stream>>>(
            rays, aplanes, alines, basisW, a_in, ptBase);
        gemm_relu_kernel<K1_PAD><<<CHUNK_PTS / 128, 256, 0, stream>>>(
            a_in, w1hT, b1, h1);
        gemm_relu_kernel<FEAT_C><<<CHUNK_PTS / 128, 256, 0, stream>>>(
            h1, w2hT, b2, h2);
        final_kernel<<<CHUNK_RAYS, 256, 0, stream>>>(
            h2, W3, b3, wgt, out, rayBase);
    }
}
